// TTTMLP_78091095376126
// MI455X (gfx1250) — matrix operand order repaired and checked by reading, not compiled, not run on hardware
//
#include <hip/hip_runtime.h>
#include <hip/hip_bf16.h>
#include <math.h>

typedef __attribute__((ext_vector_type(16))) _Float16 v16h;
typedef __attribute__((ext_vector_type(8)))  float    v8f;

#define B_N   2048
#define D_N   1024
#define R_N   64
#define H_N   256
#define OUT_N 1024
#define NPROJ 192                  // [k|v|q] concatenated columns
#define KT_PROJ (D_N / 32)         // 32 k-steps
#define NT_PROJ (NPROJ / 16)       // 12 n-tiles
#define KT_OUT  (R_N / 32)         // 2 k-steps
#define NT_OUT  (OUT_N / 16)       // 64 n-tiles
#define LN_EPS  1e-6f

// ---------------------------------------------------------------- gelu (tanh)
__device__ __forceinline__ float gelu_f(float x) {
    const float c = 0.7978845608028654f;
    float t = tanhf(c * (x + 0.044715f * x * x * x));
    return 0.5f * x * (1.f + t);
}
__device__ __forceinline__ float gelu_grad_f(float x) {
    const float c = 0.7978845608028654f;
    float x2 = x * x;
    float t = tanhf(c * (x + 0.044715f * x2 * x));
    float sech2 = 1.f - t * t;
    return 0.5f * (1.f + t) + 0.5f * x * sech2 * c * (1.f + 0.134145f * x2);
}

// --------------------------------------------------- pack Wk|Wv|Wq into WMMA-B
// B lane layout (16-bit B 32x16): lane l holds column n=l&15; K = kt*32 + (l>>4)*16 + e
__global__ void pack_b_kvq(const float* __restrict__ Wk, const float* __restrict__ Wv,
                           const float* __restrict__ Wq, _Float16* __restrict__ Bp) {
    int idx = blockIdx.x * 256 + threadIdx.x;            // [kt][nt][lane][e]
    if (idx >= KT_PROJ * NT_PROJ * 32 * 16) return;
    int e    = idx & 15;
    int lane = (idx >> 4) & 31;
    int tmp  = idx >> 9;
    int nt   = tmp % NT_PROJ;
    int kt   = tmp / NT_PROJ;
    int k = kt * 32 + ((lane >> 4) * 16) + e;
    int n = nt * 16 + (lane & 15);
    float w = (n < 64) ? Wk[k * 64 + n]
            : (n < 128) ? Wv[k * 64 + (n - 64)]
                        : Wq[k * 64 + (n - 128)];
    Bp[idx] = (_Float16)w;
}

__global__ void pack_b_wo(const float* __restrict__ Wo, _Float16* __restrict__ Bp) {
    int idx = blockIdx.x * 256 + threadIdx.x;            // [kt][nt][lane][e]
    if (idx >= KT_OUT * NT_OUT * 32 * 16) return;
    int e    = idx & 15;
    int lane = (idx >> 4) & 31;
    int tmp  = idx >> 9;
    int nt   = tmp % NT_OUT;
    int kt   = tmp / NT_OUT;
    int k = kt * 32 + ((lane >> 4) * 16) + e;
    int n = nt * 16 + (lane & 15);
    Bp[idx] = (_Float16)Wo[k * OUT_N + n];
}

// W2t[r][j] = W2[j][r]  (for coalesced dh computation)
__global__ void transpose_w2(const float* __restrict__ W2, float* __restrict__ W2t) {
    int idx = blockIdx.x * 256 + threadIdx.x;
    if (idx >= R_N * H_N) return;
    int r = idx >> 8, j = idx & 255;
    W2t[r * H_N + j] = W2[j * R_N + r];
}

// ------------------------------------------------------------ eta per sample
__global__ void eta_kernel(const float* __restrict__ x, const float* __restrict__ lr_w,
                           const float* __restrict__ lr_b, float* __restrict__ eta) {
    int b = blockIdx.x, tid = threadIdx.x;
    __shared__ float red[256];
    float s = 0.f;
    for (int i = tid; i < D_N; i += 256) s += x[(size_t)b * D_N + i] * lr_w[i];
    red[tid] = s; __syncthreads();
    for (int off = 128; off > 0; off >>= 1) {
        if (tid < off) red[tid] += red[tid + off];
        __syncthreads();
    }
    if (tid == 0) eta[b] = 0.1f / (1.f + expf(-(red[0] + lr_b[0])));
}

// ------------------------------------------- WMMA GEMM: kvq = x @ [Wk|Wv|Wq]+b
__global__ void proj_gemm(const float* __restrict__ x, const _Float16* __restrict__ Bp,
                          const float* __restrict__ bk, const float* __restrict__ bv,
                          const float* __restrict__ bq, float* __restrict__ kvq) {
    const int mtile = blockIdx.x;          // 0..127
    const int ntile = blockIdx.y;          // 0..11
    const int lane  = threadIdx.x;         // 0..31
    const int m  = lane & 15;
    const int kb = (lane >> 4) * 16;
    v8f acc = {};
    for (int kt = 0; kt < KT_PROJ; ++kt) {
        const float* arow = x + (size_t)(mtile * 16 + m) * D_N + kt * 32;
        v16h a;
#pragma unroll
        for (int e = 0; e < 8; ++e) a[e] = (_Float16)arow[kb + e];
#pragma unroll
        for (int e = 0; e < 8; ++e) a[8 + e] = (_Float16)arow[kb + 8 + e];
        const v16h bmat = *(const v16h*)(Bp + ((size_t)(kt * NT_PROJ + ntile) * 32 + lane) * 16);
        acc = __builtin_amdgcn_wmma_f32_16x16x32_f16(false, a, false, bmat,
                                                     (short)0, acc, false, false);
    }
    const int n = ntile * 16 + (lane & 15);
    const float bias = (n < 64) ? bk[n] : (n < 128) ? bv[n - 64] : bq[n - 128];
    const int m0 = (lane >> 4) * 8;
#pragma unroll
    for (int i = 0; i < 8; ++i)
        kvq[(size_t)(mtile * 16 + m0 + i) * NPROJ + n] = acc[i] + bias;
}

// ------------------------------------------------- WMMA GEMM: out = z @ Wo + bo
__global__ void out_gemm(const float* __restrict__ z, const _Float16* __restrict__ Bp,
                         const float* __restrict__ bo, float* __restrict__ out) {
    const int mtile = blockIdx.x;          // 0..127
    const int ntile = blockIdx.y;          // 0..63
    const int lane  = threadIdx.x;
    const int m  = lane & 15;
    const int kb = (lane >> 4) * 16;
    v8f acc = {};
#pragma unroll
    for (int kt = 0; kt < KT_OUT; ++kt) {
        const float* arow = z + (size_t)(mtile * 16 + m) * R_N + kt * 32;
        v16h a;
#pragma unroll
        for (int e = 0; e < 8; ++e) a[e] = (_Float16)arow[kb + e];
#pragma unroll
        for (int e = 0; e < 8; ++e) a[8 + e] = (_Float16)arow[kb + 8 + e];
        const v16h bmat = *(const v16h*)(Bp + ((size_t)(kt * NT_OUT + ntile) * 32 + lane) * 16);
        acc = __builtin_amdgcn_wmma_f32_16x16x32_f16(false, a, false, bmat,
                                                     (short)0, acc, false, false);
    }
    const int n = ntile * 16 + (lane & 15);
    const float bias = bo[n];
    const int m0 = (lane >> 4) * 8;
#pragma unroll
    for (int i = 0; i < 8; ++i)
        __builtin_nontemporal_store(acc[i] + bias,
            &out[(size_t)(mtile * 16 + m0 + i) * OUT_N + n]);
}

// ------------------------------- per-sample TTT step: rank-1 fast-weight update
__global__ void __launch_bounds__(256)
ttt_kernel(const float* __restrict__ kvq, const float* __restrict__ eta_arr,
           const float* __restrict__ W1, const float* __restrict__ W2,
           const float* __restrict__ W2t, const float* __restrict__ ln_g,
           const float* __restrict__ ln_b, float* __restrict__ W1n,
           float* __restrict__ W2n, float* __restrict__ ssl_loss,
           float* __restrict__ zbuf) {
    const int b = blockIdx.x;
    const int tid = threadIdx.x;           // 0..255, tid == j (H index)

    __shared__ float sk[R_N], sv[R_N], sq[R_N];
    __shared__ float sh[H_N], sh2[H_N];
    __shared__ float so[R_N], sxh[R_N], sdo[R_N], st64[R_N];
    __shared__ float s256[256];
    __shared__ float scal[8];

    if (tid < R_N) {
        sk[tid] = kvq[(size_t)b * NPROJ + tid];
        sv[tid] = kvq[(size_t)b * NPROJ + R_N + tid];
        sq[tid] = kvq[(size_t)b * NPROJ + 2 * R_N + tid];
    }
    __syncthreads();
    const float eta = eta_arr[b];

    // u = k.W1[:,j], uq = q.W1[:,j]   (coalesced over j)
    float u = 0.f, uq = 0.f;
#pragma unroll 8
    for (int r = 0; r < R_N; ++r) {
        float w = W1[r * H_N + tid];
        u  += sk[r] * w;
        uq += sq[r] * w;
    }
    const float h = gelu_f(u);
    sh[tid] = h;
    __syncthreads();

    // o[r] = sum_j h[j] W2[j][r] : 4 partial groups of 64 j's
    {
        int r = tid & 63, grp = tid >> 6;
        float p = 0.f;
#pragma unroll 8
        for (int jj = 0; jj < 64; ++jj) p += sh[grp * 64 + jj] * W2[(grp * 64 + jj) * R_N + r];
        s256[tid] = p;
    }
    __syncthreads();
    if (tid < R_N) so[tid] = s256[tid] + s256[64 + tid] + s256[128 + tid] + s256[192 + tid];
    __syncthreads();

    // LayerNorm(o) statistics
    if (tid == 0) {
        float s = 0.f, s2 = 0.f;
        for (int i = 0; i < R_N; ++i) { s += so[i]; s2 += so[i] * so[i]; }
        float mu = s * (1.f / R_N);
        scal[0] = mu;
        scal[1] = rsqrtf(s2 * (1.f / R_N) - mu * mu + LN_EPS);
    }
    __syncthreads();
    const float mu1 = scal[0], rstd1 = scal[1];
    if (tid < R_N) {
        float xh = (so[tid] - mu1) * rstd1;
        sxh[tid] = xh;
        float pred = sk[tid] + xh * ln_g[tid] + ln_b[tid];
        st64[tid] = pred - sv[tid];                       // diff
    }
    __syncthreads();
    if (tid == 0) {
        float s = 0.f;
        for (int i = 0; i < R_N; ++i) s += st64[i] * st64[i];
        ssl_loss[b] = s * (1.f / R_N);
    }
    __syncthreads();                                      // loss read done
    if (tid < R_N) st64[tid] = (2.f / R_N) * st64[tid] * ln_g[tid];  // dxhat
    __syncthreads();
    if (tid == 0) {
        float s = 0.f, t = 0.f;
        for (int i = 0; i < R_N; ++i) { s += st64[i]; t += st64[i] * sxh[i]; }
        scal[2] = s * (1.f / R_N);
        scal[3] = t * (1.f / R_N);
    }
    __syncthreads();
    if (tid < R_N)
        sdo[tid] = rstd1 * (st64[tid] - scal[2] - sxh[tid] * scal[3]);   // dL/do
    __syncthreads();

    // dh[j] = sum_r W2t[r][j]*do[r]; du[j] = dh*gelu'(u)   (register-resident)
    float dh = 0.f;
#pragma unroll 8
    for (int r = 0; r < R_N; ++r) dh += W2t[r * H_N + tid] * sdo[r];
    const float du = dh * gelu_grad_f(u);

    // qk = q.k
    s256[tid] = (tid < R_N) ? sq[tid] * sk[tid] : 0.f;
    __syncthreads();
    for (int off = 128; off > 0; off >>= 1) {
        if (tid < off) s256[tid] += s256[tid + off];
        __syncthreads();
    }
    const float qk = s256[0];
    __syncthreads();

    // second forward (rank-1 algebra): u2 = q.W1n[:,j] = uq - eta*qk*du
    const float u2 = uq - eta * qk * du;
    const float h2 = gelu_f(u2);
    sh2[tid] = h2;
    s256[tid] = h2 * sh[tid];                             // for hh2 = h2.h
    __syncthreads();
    for (int off = 128; off > 0; off >>= 1) {
        if (tid < off) s256[tid] += s256[tid + off];
        __syncthreads();
    }
    const float hh2 = s256[0];
    __syncthreads();

    // stream fast weights (pure rank-1 updates, nontemporal, coalesced over j)
    const size_t w1off = (size_t)b * (R_N * H_N);
#pragma unroll 4
    for (int r = 0; r < R_N; ++r)
        __builtin_nontemporal_store(W1[r * H_N + tid] - eta * sk[r] * du,
                                    &W1n[w1off + r * H_N + tid]);
    const size_t w2off = (size_t)b * (H_N * R_N);
#pragma unroll 4
    for (int c = 0; c < 64; ++c) {
        int idx = c * 256 + tid;
        int j = idx >> 6, r = idx & 63;
        __builtin_nontemporal_store(W2[idx] - eta * sh[j] * sdo[r],
                                    &W2n[w2off + idx]);
    }

    // o2[r] = h2.W2[:,r] - eta*hh2*do[r]
    {
        int r = tid & 63, grp = tid >> 6;
        float p = 0.f;
#pragma unroll 8
        for (int jj = 0; jj < 64; ++jj) p += sh2[grp * 64 + jj] * W2[(grp * 64 + jj) * R_N + r];
        s256[tid] = p;
    }
    __syncthreads();
    if (tid < R_N)
        so[tid] = s256[tid] + s256[64 + tid] + s256[128 + tid] + s256[192 + tid]
                - eta * hh2 * sdo[tid];
    __syncthreads();
    if (tid == 0) {
        float s = 0.f, s2 = 0.f;
        for (int i = 0; i < R_N; ++i) { s += so[i]; s2 += so[i] * so[i]; }
        float mu = s * (1.f / R_N);
        scal[4] = mu;
        scal[5] = rsqrtf(s2 * (1.f / R_N) - mu * mu + LN_EPS);
    }
    __syncthreads();
    if (tid < R_N) {
        float xh2 = (so[tid] - scal[4]) * scal[5];
        zbuf[(size_t)b * R_N + tid] = sq[tid] + xh2 * ln_g[tid] + ln_b[tid];
    }
}

// ---------------------------------------------------------------------- launch
extern "C" void kernel_launch(void* const* d_in, const int* in_sizes, int n_in,
                              void* d_out, int out_size, void* d_ws, size_t ws_size,
                              hipStream_t stream) {
    const float* x    = (const float*)d_in[0];
    const float* Wk   = (const float*)d_in[1];
    const float* bk   = (const float*)d_in[2];
    const float* Wv   = (const float*)d_in[3];
    const float* bv   = (const float*)d_in[4];
    const float* Wq   = (const float*)d_in[5];
    const float* bq   = (const float*)d_in[6];
    const float* Wo   = (const float*)d_in[7];
    const float* bo   = (const float*)d_in[8];
    const float* ln_g = (const float*)d_in[9];
    const float* ln_b = (const float*)d_in[10];
    const float* lr_w = (const float*)d_in[11];
    const float* lr_b = (const float*)d_in[12];
    const float* W1   = (const float*)d_in[13];
    const float* W2   = (const float*)d_in[14];

    float* out = (float*)d_out;
    float* W1n = out + (size_t)B_N * OUT_N;
    float* W2n = W1n + (size_t)B_N * R_N * H_N;
    float* ssl = W2n + (size_t)B_N * H_N * R_N;

    char* ws = (char*)d_ws;
    size_t off = 0;
    auto take = [&](size_t bytes) -> void* {
        void* p = ws + off;
        off += (bytes + 255) & ~(size_t)255;
        return p;
    };
    float*    kvq    = (float*)take((size_t)B_N * NPROJ * 4);
    float*    etab   = (float*)take((size_t)B_N * 4);
    float*    zbuf   = (float*)take((size_t)B_N * R_N * 4);
    _Float16* BpKVQ  = (_Float16*)take((size_t)KT_PROJ * NT_PROJ * 32 * 16 * 2);
    _Float16* BpWo   = (_Float16*)take((size_t)KT_OUT * NT_OUT * 32 * 16 * 2);
    float*    W2t    = (float*)take((size_t)R_N * H_N * 4);

    pack_b_kvq<<<(KT_PROJ * NT_PROJ * 32 * 16 + 255) / 256, 256, 0, stream>>>(Wk, Wv, Wq, BpKVQ);
    pack_b_wo<<<(KT_OUT * NT_OUT * 32 * 16 + 255) / 256, 256, 0, stream>>>(Wo, BpWo);
    transpose_w2<<<(R_N * H_N + 255) / 256, 256, 0, stream>>>(W2, W2t);
    eta_kernel<<<B_N, 256, 0, stream>>>(x, lr_w, lr_b, etab);
    proj_gemm<<<dim3(B_N / 16, NT_PROJ), 32, 0, stream>>>(x, BpKVQ, bk, bv, bq, kvq);
    ttt_kernel<<<B_N, 256, 0, stream>>>(kvq, etab, W1, W2, W2t, ln_g, ln_b,
                                        W1n, W2n, ssl, zbuf);
    out_gemm<<<dim3(B_N / 16, NT_OUT), 32, 0, stream>>>(zbuf, BpWo, bo, out);
}
